// PyTorchAttention_85392539779547
// MI455X (gfx1250) — matrix operand order repaired and checked by reading, not compiled, not run on hardware
//
#include <hip/hip_runtime.h>

// CDNA5 / gfx1250 flash-attention forward (causal), f16 WMMA, fp32 accumulate.
// B=4, S=2048, E=1024, H=16, d_k=64.
// KV block = 128: 36 WMMA per iteration (16 QK^T + 4 rowsum + 16 PV).
// Softmax row-state (max reduce, corr, rescale) amortized over 128 columns.
// P processed in two 64-column phases reusing one 16x64 staging buffer.

typedef __attribute__((ext_vector_type(16))) _Float16 v16h;
typedef __attribute__((ext_vector_type(8)))  _Float16 v8h;
typedef __attribute__((ext_vector_type(8)))  float    v8f;

#define B_ 4
#define S_ 2048
#define E_ 1024
#define H_ 16
#define D_ 64             // head dim
#define QTILE 16          // query rows per wave
#define KVB 128           // kv rows per block iteration
#define VPAD (KVB + 8)    // padded Vt row (16B-aligned rows)
#define WAVES 8           // waves per workgroup
#define QBLOCK (QTILE * WAVES)  // 128 query rows per workgroup

// v_exp_f32 is hardware exp2 — use the raw builtin (scores carry log2e).
__device__ __forceinline__ float fast_exp2(float x) {
    return __builtin_amdgcn_exp2f(x);
}

__device__ __forceinline__ float rowmax16(float v) {
    v = fmaxf(v, __shfl_xor(v, 1, 32));
    v = fmaxf(v, __shfl_xor(v, 2, 32));
    v = fmaxf(v, __shfl_xor(v, 4, 32));
    v = fmaxf(v, __shfl_xor(v, 8, 32));
    return v;
}

// D = A(16x32 f16) * B(32x16 f16) + C, fp32 accum
__device__ __forceinline__ v8f wmma16x16x32(v16h a, v16h b, v8f c) {
    return __builtin_amdgcn_wmma_f32_16x16x32_f16(
        false, a, false, b, (short)0, c, false, false);
}

// Load 16 contiguous halves (two 16B chunks) from LDS.
__device__ __forceinline__ v16h lds_ld16(const _Float16* p) {
    v8h lo = *(const v8h*)(p);
    v8h hi = *(const v8h*)(p + 8);
    v16h r;
#pragma unroll
    for (int j = 0; j < 8; ++j) { r[j] = lo[j]; r[j + 8] = hi[j]; }
    return r;
}

// One KV block (128 columns). MASKED = diagonal block (apply causal mask).
template <bool MASKED>
__device__ __forceinline__ void fa_block(
    int kv0, int qrow, int nc, int hi,
    const v16h& qa0, const v16h& qa1,
    const _Float16 (*Ks)[D_], const _Float16 (*Vt)[VPAD], _Float16 (*Ps)[64],
    float* m, float* l, v8f& O0, v8f& O1, v8f& O2, v8f& O3) {

    // ---- scores: eight 16x16 tiles (cols t*16..t*16+15), d=64 in two slices
    v8f s[8];
#pragma unroll
    for (int t = 0; t < 8; ++t) {
        v8f acc = {};
        v16h b0 = lds_ld16(&Ks[t * 16 + nc][hi * 16]);       // d 0..31
        v16h b1 = lds_ld16(&Ks[t * 16 + nc][32 + hi * 16]);  // d 32..63
        acc = wmma16x16x32(qa0, b0, acc);
        acc = wmma16x16x32(qa1, b1, acc);
        s[t] = acc;
    }

    // ---- per-row running max / rescale over all 128 columns ----
#pragma unroll
    for (int g = 0; g < 8; ++g) {
        if (MASKED) {
            const int r = qrow + g + 8 * hi;
#pragma unroll
            for (int t = 0; t < 8; ++t)
                if (kv0 + t * 16 + nc > r) s[t][g] = -1e30f;
        }
        float mx = fmaxf(fmaxf(fmaxf(s[0][g], s[1][g]), fmaxf(s[2][g], s[3][g])),
                         fmaxf(fmaxf(s[4][g], s[5][g]), fmaxf(s[6][g], s[7][g])));
        mx = rowmax16(mx);
        const float mnew = fmaxf(m[g], mx);
        const float corr = fast_exp2(m[g] - mnew);
        m[g] = mnew;
        l[g] *= corr;                       // row-sums added below via WMMA
        O0[g] *= corr; O1[g] *= corr; O2[g] *= corr; O3[g] *= corr;
    }

    v16h ones;
#pragma unroll
    for (int j = 0; j < 16; ++j) ones[j] = (_Float16)1.0f;

    // ---- two 64-column phases: exp -> stage P -> rowsum + PV ----
#pragma unroll
    for (int ph = 0; ph < 2; ++ph) {
#pragma unroll
        for (int g = 0; g < 8; ++g) {
            const int row = g + 8 * hi;
            Ps[row][nc]      = (_Float16)fast_exp2(s[ph * 4 + 0][g] - m[g]);
            Ps[row][16 + nc] = (_Float16)fast_exp2(s[ph * 4 + 1][g] - m[g]);
            Ps[row][32 + nc] = (_Float16)fast_exp2(s[ph * 4 + 2][g] - m[g]);
            Ps[row][48 + nc] = (_Float16)fast_exp2(s[ph * 4 + 3][g] - m[g]);
        }
        // P (C-layout in LDS) -> two A-layout slices; same-wave in-order DS
        v16h pa0, pa1;
        {
            v8h a0 = *(const v8h*)&Ps[nc][hi * 16];
            v8h a1 = *(const v8h*)&Ps[nc][8 + hi * 16];
            v8h a2 = *(const v8h*)&Ps[nc][32 + hi * 16];
            v8h a3 = *(const v8h*)&Ps[nc][40 + hi * 16];
#pragma unroll
            for (int j = 0; j < 8; ++j) {
                pa0[j] = a0[j]; pa0[j + 8] = a1[j];
                pa1[j] = a2[j]; pa1[j + 8] = a3[j];
            }
        }
        // row sums of this P half via the matrix pipe
        {
            v8f ls = {};
            ls = wmma16x16x32(pa0, ones, ls);
            ls = wmma16x16x32(pa1, ones, ls);
#pragma unroll
            for (int g = 0; g < 8; ++g) l[g] += ls[g];
        }
        // O += P[16x64] * V[64x64] for this phase's kv rows
        const int kb = ph * 64;
#pragma unroll
        for (int td = 0; td < 4; ++td) {
            v16h vb0 = lds_ld16(&Vt[td * 16 + nc][kb + hi * 16]);       // k 0..31
            v16h vb1 = lds_ld16(&Vt[td * 16 + nc][kb + 32 + hi * 16]);  // k 32..63
            v8f acc = (td == 0) ? O0 : (td == 1) ? O1 : (td == 2) ? O2 : O3;
            acc = wmma16x16x32(pa0, vb0, acc);
            acc = wmma16x16x32(pa1, vb1, acc);
            if (td == 0) O0 = acc; else if (td == 1) O1 = acc;
            else if (td == 2) O2 = acc; else O3 = acc;
        }
    }
}

__global__ __launch_bounds__(256, 2)
void fa_fwd_causal(const float* __restrict__ Q, const float* __restrict__ K,
                   const float* __restrict__ V, float* __restrict__ out) {
    __shared__ _Float16 Ks[KVB][D_];           // 16 KB, row-major K block
    __shared__ _Float16 Vt[D_][VPAD];          // 17 KB, transposed V block
    __shared__ _Float16 Ps[WAVES][QTILE][64];  // 16 KB, per-wave P staging (reused)

    const int tid  = threadIdx.x;
    const int w    = tid >> 5;
    const int lane = tid & 31;
    const int hi   = lane >> 4;
    const int nc   = lane & 15;

    const int bh = blockIdx.y;
    const int b  = bh >> 4;
    const int h  = bh & 15;
    const int qtile0 = blockIdx.x * QBLOCK;
    const int qrow   = qtile0 + w * QTILE;

    // ---- Q tile -> A registers; fold softmax scale * log2(e) into the convert
    const float qscale = 0.125f * 1.44269504088896340736f;  // 1/sqrt(64) * log2e
    const float* qp = Q + ((size_t)(b * S_ + qrow + nc)) * E_ + h * D_;
    v16h qa0, qa1;
#pragma unroll
    for (int j = 0; j < 8; ++j) {
        qa0[j]     = (_Float16)(qp[hi * 16 + j] * qscale);
        qa0[j + 8] = (_Float16)(qp[8 + hi * 16 + j] * qscale);
        qa1[j]     = (_Float16)(qp[32 + hi * 16 + j] * qscale);
        qa1[j + 8] = (_Float16)(qp[40 + hi * 16 + j] * qscale);
    }

    float m[8], l[8];
    v8f O0 = {}, O1 = {}, O2 = {}, O3 = {};
#pragma unroll
    for (int g = 0; g < 8; ++g) { m[g] = -1e30f; l[g] = 0.0f; }

    const int kv_end = qtile0 + QBLOCK;   // workgroup causal bound (<= S_)
    const int kv0_d  = qrow & ~(KVB - 1); // this wave's single diagonal block

    for (int kv0 = 0; kv0 < kv_end; kv0 += KVB) {
        __syncthreads();
        // ---- cooperative staging: 8192 f32 each of K and V, 32 per thread ----
        {
            const int krow = tid >> 1;          // 0..127
            const int d0   = (tid & 1) * 32;    // {0,32}
            const float* kg = K + ((size_t)(b * S_ + kv0 + krow)) * E_ + h * D_ + d0;
            const float* vg = V + ((size_t)(b * S_ + kv0 + krow)) * E_ + h * D_ + d0;
#pragma unroll
            for (int q = 0; q < 4; ++q) {
                v8h kk;
#pragma unroll
                for (int j = 0; j < 8; ++j) kk[j] = (_Float16)kg[q * 8 + j];
                *(v8h*)&Ks[krow][d0 + q * 8] = kk;
            }
#pragma unroll
            for (int j = 0; j < 32; ++j) Vt[d0 + j][krow] = (_Float16)vg[j];
            if (kv0 + KVB < kv_end) {
                __builtin_prefetch(kg + (size_t)KVB * E_, 0, 0);
                __builtin_prefetch(vg + (size_t)KVB * E_, 0, 0);
            }
        }
        __syncthreads();

        if (kv0 < kv0_d) {
            fa_block<false>(kv0, qrow, nc, hi, qa0, qa1, Ks, Vt, Ps[w],
                            m, l, O0, O1, O2, O3);
        } else if (kv0 == kv0_d) {
            fa_block<true>(kv0, qrow, nc, hi, qa0, qa1, Ks, Vt, Ps[w],
                           m, l, O0, O1, O2, O3);
        }
    }

    // ---- normalize and write out (fp32) ----
#pragma unroll
    for (int g = 0; g < 8; ++g) {
        const float invl = 1.0f / l[g];
        const int r = qrow + g + 8 * hi;
        float* op = out + ((size_t)(b * S_ + r)) * E_ + h * D_ + nc;
        op[0]  = O0[g] * invl;
        op[16] = O1[g] * invl;
        op[32] = O2[g] * invl;
        op[48] = O3[g] * invl;
    }
}

extern "C" void kernel_launch(void* const* d_in, const int* in_sizes, int n_in,
                              void* d_out, int out_size, void* d_ws, size_t ws_size,
                              hipStream_t stream) {
    (void)in_sizes; (void)n_in; (void)out_size; (void)d_ws; (void)ws_size;
    const float* Q = (const float*)d_in[0];
    const float* K = (const float*)d_in[1];
    const float* V = (const float*)d_in[2];
    float* out = (float*)d_out;
    dim3 grid(S_ / QBLOCK, B_ * H_);   // (16, 64)
    fa_fwd_causal<<<grid, WAVES * 32, 0, stream>>>(Q, K, V, out);
}
